// CGCNNRegressor_65420941853355
// MI455X (gfx1250) — compile-verified, matrix operand order checked by reading, not run on hardware
//
#include <hip/hip_runtime.h>
#include <hip/hip_bf16.h>

// ---------------- problem constants ----------------
#define N_NODES   50000
#define N_EDGES   800000
#define N_GRAPHS  256
#define EMB       64
#define EDGE_DIM  50
#define N_CONVS   3
#define HIDDEN    128
#define VOCAB     119
#define Z_IN      178          // 2*EMB + EDGE_DIM
#define N_KCHUNK  6            // K padded to 192 = 6 x 32
#define N_NTILE   4            // 64 cols / 16
#define BN_EPS    1e-5f

#define LDS_STRIDE 200         // f16 units per A row (400B -> bank-conflict-free)
#define TILES_PER_WAVE 5       // 8 waves * 5 tiles * 16 edges = 640 edges/block
#define EDGES_PER_BLOCK 640    // 800000 / 640 = 1250 blocks exactly

// workspace layout (bytes)
#define OFF_X       0u
#define OFF_AGG     12800000u                  // N_NODES*EMB*4
#define OFF_STATS   25600000u                  // 128 floats
#define OFF_PACKED  25600512u                  // 144 frags * 512 halfs * 2B = 147456
#define OFF_PSUM    25747968u                  // 256*64 floats
#define OFF_PCNT    25813504u                  // 256 floats
// total ~25.82 MB

typedef __attribute__((ext_vector_type(2)))  _Float16 h2;
typedef __attribute__((ext_vector_type(4)))  _Float16 h4;
typedef __attribute__((ext_vector_type(8)))  _Float16 v8h;
typedef __attribute__((ext_vector_type(16))) _Float16 v16h;
typedef __attribute__((ext_vector_type(8)))  float    v8f;

__device__ __forceinline__ float sigmoid_f(float v) {
    return 1.0f / (1.0f + expf(-v));
}
__device__ __forceinline__ float softplus_f(float v) {
    // logaddexp(v, 0) — numerically stable, matches jax.nn.softplus
    return fmaxf(v, 0.0f) + log1pf(expf(-fabsf(v)));
}
__device__ __forceinline__ float silu_f(float v) {
    return v / (1.0f + expf(-v));
}

// ---------------- weight pre-pack into WMMA B-fragment layout ----------------
// frag id = layer*48 + mat*24 + chunk*4 + tile ; each frag = 32 lanes * 16 halfs.
// B fragment layout (mirror of A): lane holds col n = tile*16 + (lane&15);
// kbase = (lane>>4)*8; element j -> K = chunk*32 + (j<8 ? kbase+j : 16+kbase+(j-8))
__global__ void pack_weights_k(const float* __restrict__ Wf,
                               const float* __restrict__ Ws,
                               _Float16* __restrict__ packed) {
    int frag  = blockIdx.x;            // 0..143
    int tile  = frag & 3;
    int chunk = (frag >> 2) % N_KCHUNK;
    int mat   = ((frag >> 2) / N_KCHUNK) & 1;
    int layer = frag / 48;
    const float* W = (mat == 0 ? Wf : Ws) + (size_t)layer * Z_IN * EMB;
    int lane = threadIdx.x;            // 32 threads
    int n  = tile * 16 + (lane & 15);
    int kb = (lane >> 4) * 8;
    _Float16* outp = packed + (size_t)frag * 512 + lane * 16;
#pragma unroll
    for (int j = 0; j < 16; ++j) {
        int kg = chunk * 32 + ((j < 8) ? (kb + j) : (16 + kb + (j - 8)));
        float v = (kg < Z_IN) ? W[(size_t)kg * EMB + n] : 0.0f;
        outp[j] = (_Float16)v;
    }
}

// ---------------- x init: embedding gather ----------------
__global__ void init_x_k(const int* __restrict__ z, const float* __restrict__ emb,
                         float* __restrict__ x) {
    int i = blockIdx.x * blockDim.x + threadIdx.x;     // N_NODES*EMB
    if (i >= N_NODES * EMB) return;
    int n = i >> 6, c = i & 63;
    int zz = z[n];
    zz = zz < 0 ? 0 : (zz > VOCAB - 1 ? VOCAB - 1 : zz);
    x[i] = emb[(size_t)zz * EMB + c];
}

__global__ void zero_f32_k(float* __restrict__ p, int n) {
    int i = blockIdx.x * blockDim.x + threadIdx.x;
    if (i < n) p[i] = 0.0f;
}

// ---------------- the hot kernel: CGConv edge GEMM via WMMA ----------------
// block = 256 threads = 8 waves; B staged once per block in LDS; each wave
// runs 5 sixteen-edge tiles against LDS-resident A and B fragments.
__global__ __launch_bounds__(256)
void edge_gemm_k(const float* __restrict__ x,
                 const int* __restrict__ srcIdx, const int* __restrict__ dstIdx,
                 const float* __restrict__ ea,
                 const _Float16* __restrict__ packed,
                 const float* __restrict__ bf, const float* __restrict__ bs,
                 float* __restrict__ agg, int layer) {
    __shared__ _Float16 aS[8 * 16 * LDS_STRIDE];   // 51200 B : per-wave A tiles
    __shared__ _Float16 bS[48 * 512];              // 49152 B : all B fragments

    const int tid  = threadIdx.x;
    const int w    = tid >> 5;
    const int lane = tid & 31;
    const int r    = lane & 15;        // A row / C column within tile
    const int p    = lane >> 4;        // half select

    // ---- stage all 48 B fragments of this layer into LDS (once per block)
    {
        const uint4* src = reinterpret_cast<const uint4*>(packed + (size_t)layer * 48 * 512);
        uint4* dstp = reinterpret_cast<uint4*>(bS);
#pragma unroll
        for (int i = 0; i < 12; ++i)               // 12 * 256 * 16B = 49152 B
            dstp[tid + i * 256] = src[tid + i * 256];
    }
    __syncthreads();

    // ---- hoist per-lane bias values (independent of tile)
    float bfv[N_NTILE], bsv[N_NTILE];
#pragma unroll
    for (int t = 0; t < N_NTILE; ++t) {
        bfv[t] = bf[layer * EMB + t * 16 + r];
        bsv[t] = bs[layer * EMB + t * 16 + r];
    }

    _Float16* arow = aS + (w * 16 + r) * LDS_STRIDE;
    const _Float16* abase = arow + p * 8;
    const _Float16* bbase = bS + lane * 16;

    for (int it = 0; it < TILES_PER_WAVE; ++it) {
        const int tileBase = blockIdx.x * EDGES_PER_BLOCK + (it * 8 + w) * 16;

        // ---- gather ze = [x[dst](64) | x[src](64) | edge_attr(50) | pad(14)]
        {
            const int e = tileBase + r;
            const int s = srcIdx[e];
            const int d = dstIdx[e];
            if (p == 0) {
                const float4* xd = reinterpret_cast<const float4*>(x + (size_t)d * EMB);
#pragma unroll
                for (int k = 0; k < 16; ++k) {
                    float4 v = xd[k];
                    h4 t = { (_Float16)v.x, (_Float16)v.y, (_Float16)v.z, (_Float16)v.w };
                    *reinterpret_cast<h4*>(arow + 4 * k) = t;
                }
                const float4* xs = reinterpret_cast<const float4*>(x + (size_t)s * EMB);
#pragma unroll
                for (int k = 0; k < 8; ++k) {
                    float4 v = xs[k];
                    h4 t = { (_Float16)v.x, (_Float16)v.y, (_Float16)v.z, (_Float16)v.w };
                    *reinterpret_cast<h4*>(arow + 64 + 4 * k) = t;
                }
            } else {
                const float4* xs = reinterpret_cast<const float4*>(x + (size_t)s * EMB);
#pragma unroll
                for (int k = 8; k < 16; ++k) {
                    float4 v = xs[k];
                    h4 t = { (_Float16)v.x, (_Float16)v.y, (_Float16)v.z, (_Float16)v.w };
                    *reinterpret_cast<h4*>(arow + 64 + 4 * k) = t;
                }
                const float2* e2 = reinterpret_cast<const float2*>(ea + (size_t)e * EDGE_DIM);
#pragma unroll
                for (int k = 0; k < 25; ++k) {
                    float2 v = e2[k];
                    h2 t = { (_Float16)v.x, (_Float16)v.y };
                    *reinterpret_cast<h2*>(arow + 128 + 2 * k) = t;
                }
                h2 zz = { (_Float16)0.0f, (_Float16)0.0f };
#pragma unroll
                for (int k = 0; k < 7; ++k)
                    *reinterpret_cast<h2*>(arow + 178 + 2 * k) = zz;
            }
        }
        // wave-local sync: LDS ops are in-order per wave; wait our stores,
        // forbid compiler from hoisting the A loads above them.
        asm volatile("s_wait_dscnt 0x0" ::: "memory");

        // ---- load all 6 A fragments for this tile (ds_load_b128 pairs)
        v16h a[N_KCHUNK];
#pragma unroll
        for (int c = 0; c < N_KCHUNK; ++c) {
            *reinterpret_cast<v8h*>(&a[c])       = *reinterpret_cast<const v8h*>(abase + c * 32);
            *(reinterpret_cast<v8h*>(&a[c]) + 1) = *reinterpret_cast<const v8h*>(abase + c * 32 + 16);
        }

        // ---- 48 WMMAs against LDS-resident B fragments
        v8f accF[N_NTILE];
        v8f accS[N_NTILE];
#pragma unroll
        for (int t = 0; t < N_NTILE; ++t) { accF[t] = (v8f){}; accS[t] = (v8f){}; }

#pragma unroll
        for (int c = 0; c < N_KCHUNK; ++c) {
#pragma unroll
            for (int t = 0; t < N_NTILE; ++t) {
                const _Float16* pf = bbase + (size_t)(0 * 24 + c * 4 + t) * 512;
                const _Float16* ps = bbase + (size_t)(1 * 24 + c * 4 + t) * 512;
                v16h bF, bSv;
                *reinterpret_cast<v8h*>(&bF)       = *reinterpret_cast<const v8h*>(pf);
                *(reinterpret_cast<v8h*>(&bF) + 1) = *reinterpret_cast<const v8h*>(pf + 8);
                *reinterpret_cast<v8h*>(&bSv)       = *reinterpret_cast<const v8h*>(ps);
                *(reinterpret_cast<v8h*>(&bSv) + 1) = *reinterpret_cast<const v8h*>(ps + 8);
                accF[t] = __builtin_amdgcn_wmma_f32_16x16x32_f16(
                    false, a[c], false, bF, (short)0, accF[t], false, false);
                accS[t] = __builtin_amdgcn_wmma_f32_16x16x32_f16(
                    false, a[c], false, bSv, (short)0, accS[t], false, false);
            }
        }

        // ---- epilogue: m = sigmoid(f+bf) * softplus(s+bs) -> atomic scatter
        int dsts[8];
#pragma unroll
        for (int rr = 0; rr < 8; ++rr)
            dsts[rr] = dstIdx[tileBase + p * 8 + rr];

#pragma unroll
        for (int t = 0; t < N_NTILE; ++t) {
            const int n = t * 16 + r;              // output channel
#pragma unroll
            for (int rr = 0; rr < 8; ++rr) {       // C/D row m = p*8 + rr
                float g  = sigmoid_f(accF[t][rr] + bfv[t]);
                float sp = softplus_f(accS[t][rr] + bsv[t]);
                atomicAdd(agg + (size_t)dsts[rr] * EMB + n, g * sp);
            }
        }

        // hardware keeps per-wave LDS ops in order (loads above complete
        // before next iteration's stores); just stop compiler reordering.
        asm volatile("" ::: "memory");
    }
}

// ---------------- BatchNorm statistics (sum / sumsq per channel) ----------------
__global__ void bn_stats_k(const float* __restrict__ agg, float* __restrict__ stats) {
    int tid   = blockIdx.x * blockDim.x + threadIdx.x;   // 65536 threads
    int ch    = tid & 63;
    int slice = tid >> 6;                                // 0..1023
    float s = 0.0f, s2 = 0.0f;
    for (int n = slice; n < N_NODES; n += 1024) {
        float v = agg[(size_t)n * EMB + ch];
        s += v; s2 += v * v;
    }
    atomicAdd(&stats[ch], s);
    atomicAdd(&stats[64 + ch], s2);
}

// ---------------- x = silu(x + BN(agg)) ----------------
__global__ void bn_update_k(float* __restrict__ x, const float* __restrict__ agg,
                            const float* __restrict__ stats,
                            const float* __restrict__ gamma, const float* __restrict__ beta,
                            int layer) {
    int i = blockIdx.x * blockDim.x + threadIdx.x;       // N_NODES*EMB
    if (i >= N_NODES * EMB) return;
    int ch = i & 63;
    const float inv = 1.0f / (float)N_NODES;
    float mu  = stats[ch] * inv;
    float var = stats[64 + ch] * inv - mu * mu;
    float bn  = gamma[layer * EMB + ch] * (agg[i] - mu) * rsqrtf(var + BN_EPS)
              + beta[layer * EMB + ch];
    x[i] = silu_f(x[i] + bn);
}

// ---------------- global mean pool (accumulate) ----------------
__global__ void pool_k(const float* __restrict__ x, const int* __restrict__ batch,
                       float* __restrict__ sums, float* __restrict__ counts) {
    int i = blockIdx.x * blockDim.x + threadIdx.x;       // N_NODES*EMB
    if (i >= N_NODES * EMB) return;
    int n = i >> 6, ch = i & 63;
    int g = batch[n];
    atomicAdd(&sums[(size_t)g * EMB + ch], x[i]);
    if (ch == 0) atomicAdd(&counts[g], 1.0f);
}

// ---------------- per-graph MLP head: 64 -> 128 -> 1 ----------------
__global__ void mlp_k(const float* __restrict__ sums, const float* __restrict__ counts,
                      const float* __restrict__ W1, const float* __restrict__ b1,
                      const float* __restrict__ W2, const float* __restrict__ b2,
                      float* __restrict__ out) {
    __shared__ float gg[EMB];
    __shared__ float red[HIDDEN];
    int g = blockIdx.x, t = threadIdx.x;                 // 128 threads
    float cnt = fmaxf(counts[g], 1.0f);
    if (t < EMB) gg[t] = sums[(size_t)g * EMB + t] / cnt;
    __syncthreads();
    float acc = b1[t];
#pragma unroll
    for (int k = 0; k < EMB; ++k) acc += gg[k] * W1[k * HIDDEN + t];
    red[t] = silu_f(acc) * W2[t];
    __syncthreads();
    for (int sft = HIDDEN / 2; sft > 0; sft >>= 1) {
        if (t < sft) red[t] += red[t + sft];
        __syncthreads();
    }
    if (t == 0) out[g] = red[0] + b2[0];
}

// ---------------- launch ----------------
extern "C" void kernel_launch(void* const* d_in, const int* in_sizes, int n_in,
                              void* d_out, int out_size, void* d_ws, size_t ws_size,
                              hipStream_t stream) {
    const int*   z     = (const int*)d_in[0];
    const int*   eidx  = (const int*)d_in[1];     // [2, E]: row0=src, row1=dst
    const float* ea    = (const float*)d_in[2];
    const int*   batch = (const int*)d_in[3];
    const float* emb   = (const float*)d_in[4];
    const float* Wf    = (const float*)d_in[5];
    const float* bf    = (const float*)d_in[6];
    const float* Ws    = (const float*)d_in[7];
    const float* bs    = (const float*)d_in[8];
    const float* gamma = (const float*)d_in[9];
    const float* beta  = (const float*)d_in[10];
    const float* W1    = (const float*)d_in[11];
    const float* b1    = (const float*)d_in[12];
    const float* W2    = (const float*)d_in[13];
    const float* b2    = (const float*)d_in[14];
    float* out = (float*)d_out;

    char* ws = (char*)d_ws;
    float*    x      = (float*)(ws + OFF_X);
    float*    agg    = (float*)(ws + OFF_AGG);
    float*    stats  = (float*)(ws + OFF_STATS);
    _Float16* packed = (_Float16*)(ws + OFF_PACKED);
    float*    psum   = (float*)(ws + OFF_PSUM);
    float*    pcnt   = (float*)(ws + OFF_PCNT);

    const int* srcIdx = eidx;
    const int* dstIdx = eidx + N_EDGES;

    pack_weights_k<<<N_CONVS * 2 * N_KCHUNK * N_NTILE, 32, 0, stream>>>(Wf, Ws, packed);
    init_x_k<<<(N_NODES * EMB) / 256, 256, 0, stream>>>(z, emb, x);

    for (int l = 0; l < N_CONVS; ++l) {
        // zero agg (3.2M) + stats (128) — contiguous in ws
        zero_f32_k<<<(N_NODES * EMB + 128 + 255) / 256, 256, 0, stream>>>(agg, N_NODES * EMB + 128);
        edge_gemm_k<<<N_EDGES / EDGES_PER_BLOCK, 256, 0, stream>>>(x, srcIdx, dstIdx, ea, packed, bf, bs, agg, l);
        bn_stats_k<<<256, 256, 0, stream>>>(agg, stats);
        bn_update_k<<<(N_NODES * EMB) / 256, 256, 0, stream>>>(x, agg, stats, gamma, beta, l);
    }

    // zero pool sums (256*64) + counts (256) — contiguous
    zero_f32_k<<<(N_GRAPHS * EMB + N_GRAPHS + 255) / 256, 256, 0, stream>>>(psum, N_GRAPHS * EMB + N_GRAPHS);
    pool_k<<<(N_NODES * EMB) / 256, 256, 0, stream>>>(x, batch, psum, pcnt);
    mlp_k<<<N_GRAPHS, HIDDEN, 0, stream>>>(psum, pcnt, W1, b1, W2, b2, out);
}
